// EntropyConv_80453327388793
// MI455X (gfx1250) — matrix operand order repaired and checked by reading, not compiled, not run on hardware
//
#include <hip/hip_runtime.h>

typedef __attribute__((ext_vector_type(16))) __bf16 v16bf;
typedef __attribute__((ext_vector_type(8)))  float  v8f;

#define B_     8
#define H_     256
#define W_     512
#define CI_    32
#define CO_    32
#define NPART_ 8
#define HP_    32          // rows per partition
#define RT_    16          // output rows per block
#define CT_    64          // output cols per block
#define XROWS  (RT_ + 2)   // 18 (2-row top halo only; kh<=2)
#define XCOLS  (CT_ + 4)   // 68 (+-2 col halo)
#define NTAPS  13          // live taps after PixelCNN mask
#define XT_ELEMS (XROWS * XCOLS * CI_)   // 39168 bf16
#define WT_ELEMS (NTAPS * 2 * 32 * 16)   // 13312 bf16

__global__ __launch_bounds__(256) void pixelcnn_conv_wmma(
    const float* __restrict__ x, const float* __restrict__ weight,
    const float* __restrict__ bias, const float* __restrict__ alpha,
    const int* __restrict__ widths, float* __restrict__ out)
{
  extern __shared__ __bf16 lds[];
  __bf16* xt = lds;               // x tile, [row][col][ci-swizzled]
  __bf16* wt = lds + XT_ELEMS;    // masked weights in WMMA A-operand striping

  const int tid = threadIdx.x;
  int bid = blockIdx.x;
  const int ct   = bid & 7;  bid >>= 3;   // col strip (W/CT = 8)
  const int rt   = bid & 1;  bid >>= 1;   // row half of partition (HP/RT = 2)
  const int part = bid & 7;  bid >>= 3;
  const int b    = bid;

  const int c0    = ct * CT_;
  const int r0    = rt * RT_;
  const int hbase = part * HP_;
  const int width = widths[part];

  // ---- phase 1: masked weights -> LDS, A-operand (16x32 bf16) striping ----
  // live taps t=0..12: kh = t/5, kw = t%5  (covers (0,0..4),(1,0..4),(2,0..2))
  for (int i = tid; i < NTAPS * CO_ * CI_; i += 256) {
    int t   = i >> 10;         // / (32*32)
    int rem = i & 1023;
    int co  = rem >> 5;
    int ci  = rem & 31;
    int kh  = t / 5;
    int kw  = t - kh * 5;
    float wv = weight[((co * CI_ + ci) * 5 + kh) * 5 + kw];
    if (kh == 2 && kw == 2 && ((ci >> 2) > (co >> 2))) wv = 0.f;  // center tap: gin<=gout
    // A operand in this code's contiguous k order (same as B): lane = M + 16*bit4(K); slot = K&15
    int lane = (co & 15) + ((ci & 16) ? 16 : 0);
    int slot = ci & 15;
    wt[((t * 2 + (co >> 4)) * 32 + lane) * 16 + slot] = (__bf16)wv;
  }

  // ---- phase 2: x tile -> LDS bf16 ----
  // Each thread: 4 consecutive ci at one (row, col). Global: 4 lane-coalesced
  // 128B b32 loads. LDS: one contiguous 8B ds_store_b64. Odd columns swap the
  // two 32B ci-halves (swizzle) to spread store banks; compute reads adjust.
  for (int i = tid; i < XROWS * 8 * XCOLS; i += 256) {
    int col = i % XCOLS;                    // col innermost -> coalesced global
    int r2  = i / XCOLS;
    int cig = r2 & 7;                       // ci group of 4 (8 groups)
    int row = r2 >> 3;
    int gc = c0 + col - 2;
    int gr = r0 + row - 2;                  // partition-local row
    float v0 = 0.f, v1 = 0.f, v2 = 0.f, v3 = 0.f;
    if (gc >= 0 && gr >= 0 && gc < width) { // width <= 502 < W, also bounds W
      const float* xp = &x[((b * CI_ + (cig << 2)) * H_ + hbase + gr) * W_ + gc];
      v0 = xp[0 * H_ * W_];
      v1 = xp[1 * H_ * W_];
      v2 = xp[2 * H_ * W_];
      v3 = xp[3 * H_ * W_];
    }
    union { __bf16 h[4]; uint2 u; } pk;
    pk.h[0] = (__bf16)v0; pk.h[1] = (__bf16)v1;
    pk.h[2] = (__bf16)v2; pk.h[3] = (__bf16)v3;
    int slot = ((cig << 2) + ((col & 1) << 4)) & 31;   // half-swap swizzle
    *(uint2*)&xt[(row * XCOLS + col) * CI_ + slot] = pk.u;
  }
  __syncthreads();

  const int wave     = tid >> 5;
  const int lane     = tid & 31;
  const int laneN    = lane & 15;
  const int laneHalf = lane >> 4;

  float biasA[8], alphaA[8], biasB[8], alphaB[8];
  #pragma unroll
  for (int i = 0; i < 8; ++i) {
    int co = i + 8 * laneHalf;              // C/D layout: VGPR i -> M = i + 8*(lane>=16)
    biasA[i] = bias[co];       alphaA[i] = alpha[co];
    biasB[i] = bias[co + 16];  alphaB[i] = alpha[co + 16];
  }

  // ---- phase 3: 8 (16px x 32co) tiles per wave, 26 WMMAs each ----
  for (int t8 = 0; t8 < 8; ++t8) {
    int tile  = wave * 8 + t8;              // 64 tiles: 16 rows x 4 col-tiles
    int row   = tile & 15;
    int ctile = tile >> 4;
    v8f acc0 = {};
    v8f acc1 = {};
    #pragma unroll
    for (int t = 0; t < NTAPS; ++t) {
      const int kh = t / 5, kw = t - kh * 5;
      int lrow = row + kh;                          // (r0+row)+(kh-2) in padded tile
      int lcol = ctile * 16 + laneN + kw;           // lane's pixel column + shift
      // B operand 32x16: lanes 0-15 ci 0-15, lanes 16-31 ci 16-31 (un-swizzled);
      // chunk base follows the col-parity half-swap, still one contiguous 32B read
      int chunk = (laneHalf ^ (lcol & 1)) << 4;
      const v16bf bmat = *(const v16bf*)&xt[(lrow * XCOLS + lcol) * CI_ + chunk];
      const v16bf a0   = *(const v16bf*)&wt[((t * 2 + 0) * 32 + lane) * 16];
      const v16bf a1   = *(const v16bf*)&wt[((t * 2 + 1) * 32 + lane) * 16];
      acc0 = __builtin_amdgcn_wmma_f32_16x16x32_bf16(false, a0, false, bmat,
                                                     (short)0, acc0, false, false);
      acc1 = __builtin_amdgcn_wmma_f32_16x16x32_bf16(false, a1, false, bmat,
                                                     (short)0, acc1, false, false);
    }
    int gh = hbase + r0 + row;
    int gc = c0 + ctile * 16 + laneN;
    float validf = (gc < width) ? 1.f : 0.f;
    #pragma unroll
    for (int i = 0; i < 8; ++i) {
      int coA = i + 8 * laneHalf;
      int coB = coA + 16;
      float y0 = acc0[i] + biasA[i];
      y0 = (y0 > 0.f) ? y0 : alphaA[i] * y0;
      float y1 = acc1[i] + biasB[i];
      y1 = (y1 > 0.f) ? y1 : alphaB[i] * y1;
      out[((b * CO_ + coA) * H_ + gh) * W_ + gc] = y0 * validf;
      out[((b * CO_ + coB) * H_ + gh) * W_ + gc] = y1 * validf;
    }
  }
}

extern "C" void kernel_launch(void* const* d_in, const int* in_sizes, int n_in,
                              void* d_out, int out_size, void* d_ws, size_t ws_size,
                              hipStream_t stream) {
  const float* x      = (const float*)d_in[0];
  const float* weight = (const float*)d_in[1];
  const float* bias   = (const float*)d_in[2];
  const float* alpha  = (const float*)d_in[3];
  const int*   widths = (const int*)d_in[4];
  float* out = (float*)d_out;

  const size_t lds_bytes = (size_t)(XT_ELEMS + WT_ELEMS) * 2;  // ~105 KB dynamic LDS
  dim3 grid(B_ * NPART_ * (HP_ / RT_) * (W_ / CT_));           // 1024 blocks
  pixelcnn_conv_wmma<<<grid, 256, lds_bytes, stream>>>(x, weight, bias, alpha,
                                                       widths, out);
}
